// Encoder_83348135346454
// MI455X (gfx1250) — matrix operand order repaired and checked by reading, not compiled, not run on hardware
//
#include <hip/hip_runtime.h>
#include <hip/hip_bf16.h>

// Problem sizes (fixed by the reference)
#define BB 1024
#define TT 64
#define DD 256
#define HH 256
#define KDIM 512      // D + H
#define N4H 1024      // 4*H

typedef __attribute__((ext_vector_type(16))) __bf16 v16bf;
typedef __attribute__((ext_vector_type(8)))  float  v8f;

__device__ __forceinline__ unsigned short f2bf(float f) {
  unsigned int u = __float_as_uint(f);
  u += 0x7FFFu + ((u >> 16) & 1u);      // round-to-nearest-even
  return (unsigned short)(u >> 16);
}
__device__ __forceinline__ float sigmoidf_(float x) {
  return 1.0f / (1.0f + __expf(-x));
}
// A-fragment (16x32 bf16) packed ushort index for element (row m, k), per tile
__device__ __forceinline__ int afrag_idx(int m, int k) {
  int kk = k & 31;
  return (k >> 5) * 512 + (m + 16 * ((kk >> 3) & 1)) * 16
       + (((kk & 7) >> 1) + 4 * (kk >> 4)) * 2 + (kk & 1);
}

// ---------------------------------------------------------------------------
// Pack W = [W_ih | W_hh]  (N=1024 x K=512) into bf16 B-fragment layout:
// Wp[ktile][ntile][lane][16 bf16] -> one contiguous 32B load per lane.
// ---------------------------------------------------------------------------
__global__ __launch_bounds__(256) void enc_pack_w(
    const float* __restrict__ Wih, const float* __restrict__ Whh,
    unsigned short* __restrict__ Wp)
{
  int idx = blockIdx.x * 256 + threadIdx.x;     // 0 .. 1024*512-1
  int n = idx >> 9;
  int k = idx & 511;
  float v = (k < 256) ? Wih[n * 256 + k] : Whh[n * 256 + (k - 256)];
  int kt = k >> 5, kk = k & 31, ntile = n >> 4, nn = n & 15;
  int lane = ((kk >> 3) & 1) * 16 + nn;
  int vg = ((kk & 7) >> 1) + 4 * (kk >> 4), pr = kk & 1;
  Wp[((size_t)(kt * 64 + ntile) * 32 + lane) * 16 + vg * 2 + pr] = f2bf(v);
}

__global__ __launch_bounds__(256) void enc_bias(
    const float* __restrict__ bih, const float* __restrict__ bhh,
    float* __restrict__ biasC)
{
  int n = blockIdx.x * 256 + threadIdx.x;
  if (n < N4H) biasC[n] = bih[n] + bhh[n];
}

// ---------------------------------------------------------------------------
// Attention (softmax is shift-invariant => independent of h,c) and
// input_weighted = attn * x_t.  Also emits w_in pre-packed as bf16
// A-fragments (8KB per (btile,t)) for async staging in the recurrent kernel.
// ---------------------------------------------------------------------------
__global__ __launch_bounds__(256) void enc_attn_win(
    const float* __restrict__ x,       // [B][T][D]
    const float* __restrict__ Wattn,   // [2H+T]
    const float* __restrict__ battn,   // [1]
    float* __restrict__ outW,          // [B][T][D]
    unsigned short* __restrict__ Apre) // packed bf16 w_in, or nullptr
{
  __shared__ float red[256];
  const int b = blockIdx.x;
  const int d = threadIdx.x;
  float xv[TT];
  float s = battn[0];
#pragma unroll
  for (int t = 0; t < TT; ++t) {
    xv[t] = x[((size_t)b * TT + t) * DD + d];
    s += xv[t] * Wattn[2 * HH + t];             // w_x uniform -> s_loads
  }
  red[d] = s; __syncthreads();
  for (int off = 128; off > 0; off >>= 1) {
    if (d < off) red[d] = fmaxf(red[d], red[d + off]);
    __syncthreads();
  }
  float mx = red[0]; __syncthreads();
  float p = __expf(s - mx);
  red[d] = p; __syncthreads();
  for (int off = 128; off > 0; off >>= 1) {
    if (d < off) red[d] += red[d + off];
    __syncthreads();
  }
  float attn = p / red[0];
#pragma unroll
  for (int t = 0; t < TT; ++t)
    outW[((size_t)b * TT + t) * DD + d] = attn * xv[t];
  if (Apre) {
    const int btile = b >> 4, row = b & 15;
    const int fidx = afrag_idx(row, d);
#pragma unroll
    for (int t = 0; t < TT; ++t)
      Apre[(size_t)(btile * 64 + t) * 4096 + fidx] = f2bf(attn * xv[t]);
  }
}

// ---------------------------------------------------------------------------
// Recurrent LSTM (fast path): 64 WGs x 256 threads, 16 batch rows/WG.
// w_in tiles arrive via double-buffered global_load_async_to_lds_b128,
// issued one timestep ahead (overlaps previous step's WMMA + elementwise).
// ---------------------------------------------------------------------------
__global__ __launch_bounds__(256) void enc_recurrent_async(
    const unsigned short* __restrict__ Apre,   // packed bf16 w_in tiles
    const unsigned short* __restrict__ Wp,     // packed bf16 weights
    const float* __restrict__ biasC,           // [1024]
    float* __restrict__ outEnc)                // [B][T][H]
{
  __shared__ __align__(16) unsigned short sAlo[2][4096]; // w_in frags (2x8KB)
  __shared__ __align__(32) unsigned short sAh[4096];     // h frags (8KB)
  __shared__ float sG[16 * N4H];                         // 64KB gates

  const int tid   = threadIdx.x;
  const int mbase = blockIdx.x * 16;
  const int wave  = tid >> 5;
  const int lane  = tid & 31;
  const int em    = tid >> 4;
  const int ej0   = tid & 15;

  // kick off async copy of t=0 tile into buffer 0 (32B per thread)
  {
    const unsigned short* gsrc = Apre + (size_t)(blockIdx.x * 64 + 0) * 4096 + tid * 8;
    unsigned ldst = (unsigned)(size_t)&sAlo[0][tid * 8];
    asm volatile("global_load_async_to_lds_b128 %0, %1, off"
                 :: "v"(ldst), "v"(gsrc) : "memory");
    asm volatile("global_load_async_to_lds_b128 %0, %1, off offset:4096"
                 :: "v"(ldst), "v"(gsrc) : "memory");
  }

  // zero h fragments
  for (int i = tid; i < 4096; i += 256) sAh[i] = 0;

  float cReg[16];
#pragma unroll
  for (int r = 0; r < 16; ++r) cReg[r] = 0.0f;

  float bn[8];
#pragma unroll
  for (int q = 0; q < 8; ++q)
    bn[q] = biasC[(wave * 8 + q) * 16 + (lane & 15)];

  for (int t = 0; t < TT; ++t) {
    // prefetch next step's w_in tile into the other buffer, wait for current
    if (t + 1 < TT) {
      const unsigned short* gsrc =
          Apre + (size_t)(blockIdx.x * 64 + (t + 1)) * 4096 + tid * 8;
      unsigned ldst = (unsigned)(size_t)&sAlo[(t + 1) & 1][tid * 8];
      asm volatile("global_load_async_to_lds_b128 %0, %1, off"
                   :: "v"(ldst), "v"(gsrc) : "memory");
      asm volatile("global_load_async_to_lds_b128 %0, %1, off offset:4096"
                   :: "v"(ldst), "v"(gsrc) : "memory");
      asm volatile("s_wait_asynccnt 0x2" ::: "memory");   // step-t ops done
    } else {
      asm volatile("s_wait_asynccnt 0x0" ::: "memory");
    }
    __syncthreads();

    // --- WMMA: M=16, N = wave's 8 tiles of 16, K = 512
    v8f acc[8];
#pragma unroll
    for (int q = 0; q < 8; ++q) {
      v8f a0;
#pragma unroll
      for (int e = 0; e < 8; ++e) a0[e] = bn[q];
      acc[q] = a0;
    }
#pragma unroll 2
    for (int kt = 0; kt < 8; ++kt) {          // K = 0..255 (w_in)
      const v16bf a = *reinterpret_cast<const v16bf*>(
          &sAlo[t & 1][kt * 512 + lane * 16]);
      __builtin_prefetch(&Wp[((size_t)((kt + 1) * 64 + wave * 8) * 32 + lane) * 16], 0, 3);
#pragma unroll
      for (int q = 0; q < 8; ++q) {
        int nt = wave * 8 + q;
        const v16bf b = *reinterpret_cast<const v16bf*>(
            &Wp[((size_t)(kt * 64 + nt) * 32 + lane) * 16]);
        acc[q] = __builtin_amdgcn_wmma_f32_16x16x32_bf16(
            false, a, false, b, (short)0, acc[q], false, false);
      }
    }
#pragma unroll 2
    for (int kt = 8; kt < 16; ++kt) {         // K = 256..511 (h)
      const v16bf a = *reinterpret_cast<const v16bf*>(
          &sAh[(kt - 8) * 512 + lane * 16]);
      if (kt + 1 < 16)
        __builtin_prefetch(&Wp[((size_t)((kt + 1) * 64 + wave * 8) * 32 + lane) * 16], 0, 3);
#pragma unroll
      for (int q = 0; q < 8; ++q) {
        int nt = wave * 8 + q;
        const v16bf b = *reinterpret_cast<const v16bf*>(
            &Wp[((size_t)(kt * 64 + nt) * 32 + lane) * 16]);
        acc[q] = __builtin_amdgcn_wmma_f32_16x16x32_bf16(
            false, a, false, b, (short)0, acc[q], false, false);
      }
    }
    // spill gates to LDS: lane<16 -> rows 0..7, lane>=16 -> rows 8..15
#pragma unroll
    for (int q = 0; q < 8; ++q) {
      int n0 = (wave * 8 + q) * 16 + (lane & 15);
      int rbase = (lane >> 4) * 8;
#pragma unroll
      for (int e = 0; e < 8; ++e)
        sG[(rbase + e) * N4H + n0] = acc[q][e];
    }
    __syncthreads();

    // elementwise LSTM cell, c kept in registers; h_new -> sAh + global
#pragma unroll
    for (int r = 0; r < 16; ++r) {
      int j = r * 16 + ej0;
      float gi = sG[em * N4H + j];
      float gf = sG[em * N4H + 256 + j];
      float gg = sG[em * N4H + 512 + j];
      float go = sG[em * N4H + 768 + j];
      float cn = sigmoidf_(gf) * cReg[r] + sigmoidf_(gi) * tanhf(gg);
      float hn = sigmoidf_(go) * tanhf(cn);
      cReg[r] = cn;
      outEnc[((size_t)(mbase + em) * TT + t) * HH + j] = hn;
      sAh[afrag_idx(em, j)] = f2bf(hn);   // k = 256+j, local to sAh
    }
  }
}

// ---------------------------------------------------------------------------
// Fallback recurrent kernel (no Apre workspace): stages w_in from f32 outW.
// ---------------------------------------------------------------------------
__global__ __launch_bounds__(256) void enc_recurrent(
    const float* __restrict__ outW,
    const unsigned short* __restrict__ Wp,
    const float* __restrict__ biasC,
    float* __restrict__ outEnc)
{
  __shared__ __align__(32) unsigned short sA[16 * 512];
  __shared__ float sG[16 * N4H];

  const int tid   = threadIdx.x;
  const int mbase = blockIdx.x * 16;
  const int wave  = tid >> 5;
  const int lane  = tid & 31;
  const int em    = tid >> 4;
  const int ej0   = tid & 15;

  for (int i = tid; i < 4096; i += 256) sA[4096 + i] = 0;

  float cReg[16];
#pragma unroll
  for (int r = 0; r < 16; ++r) cReg[r] = 0.0f;
  float bn[8];
#pragma unroll
  for (int q = 0; q < 8; ++q)
    bn[q] = biasC[(wave * 8 + q) * 16 + (lane & 15)];

  for (int t = 0; t < TT; ++t) {
#pragma unroll
    for (int i = 0; i < 16; ++i) {
      int flat = tid + i * 256;
      int row = flat >> 8;
      int d   = flat & 255;
      float v = outW[((size_t)(mbase + row) * TT + t) * DD + d];
      sA[afrag_idx(row, d)] = f2bf(v);
    }
    __syncthreads();

    v8f acc[8];
#pragma unroll
    for (int q = 0; q < 8; ++q) {
      v8f a0;
#pragma unroll
      for (int e = 0; e < 8; ++e) a0[e] = bn[q];
      acc[q] = a0;
    }
#pragma unroll 2
    for (int kt = 0; kt < 16; ++kt) {
      const v16bf a = *reinterpret_cast<const v16bf*>(&sA[kt * 512 + lane * 16]);
      if (kt + 1 < 16)
        __builtin_prefetch(&Wp[((size_t)((kt + 1) * 64 + wave * 8) * 32 + lane) * 16], 0, 3);
#pragma unroll
      for (int q = 0; q < 8; ++q) {
        int nt = wave * 8 + q;
        const v16bf b = *reinterpret_cast<const v16bf*>(
            &Wp[((size_t)(kt * 64 + nt) * 32 + lane) * 16]);
        acc[q] = __builtin_amdgcn_wmma_f32_16x16x32_bf16(
            false, a, false, b, (short)0, acc[q], false, false);
      }
    }
#pragma unroll
    for (int q = 0; q < 8; ++q) {
      int n0 = (wave * 8 + q) * 16 + (lane & 15);
      int rbase = (lane >> 4) * 8;
#pragma unroll
      for (int e = 0; e < 8; ++e)
        sG[(rbase + e) * N4H + n0] = acc[q][e];
    }
    __syncthreads();

#pragma unroll
    for (int r = 0; r < 16; ++r) {
      int j = r * 16 + ej0;
      float gi = sG[em * N4H + j];
      float gf = sG[em * N4H + 256 + j];
      float gg = sG[em * N4H + 512 + j];
      float go = sG[em * N4H + 768 + j];
      float cn = sigmoidf_(gf) * cReg[r] + sigmoidf_(gi) * tanhf(gg);
      float hn = sigmoidf_(go) * tanhf(cn);
      cReg[r] = cn;
      outEnc[((size_t)(mbase + em) * TT + t) * HH + j] = hn;
      sA[4096 + afrag_idx(em, j)] = f2bf(hn);
    }
  }
}

// ---------------------------------------------------------------------------
extern "C" void kernel_launch(void* const* d_in, const int* in_sizes, int n_in,
                              void* d_out, int out_size, void* d_ws, size_t ws_size,
                              hipStream_t stream) {
  const float* x     = (const float*)d_in[0];
  const float* Wattn = (const float*)d_in[1];
  const float* battn = (const float*)d_in[2];
  const float* Wih   = (const float*)d_in[3];
  const float* Whh   = (const float*)d_in[4];
  const float* bih   = (const float*)d_in[5];
  const float* bhh   = (const float*)d_in[6];

  float* outW   = (float*)d_out;
  float* outEnc = outW + (size_t)BB * TT * DD;

  // workspace layout: Wp (1MB) | biasC (4KB) | pad | Apre (32MB, fast path)
  const size_t offBias = (size_t)N4H * KDIM * 2;
  const size_t offApre = (offBias + 4096 + 255) & ~(size_t)255;
  const size_t needFast = offApre + (size_t)(BB / 16) * TT * 4096 * 2;
  unsigned short* Wp   = (unsigned short*)d_ws;
  float* biasC         = (float*)((char*)d_ws + offBias);
  const bool fast      = ws_size >= needFast;
  unsigned short* Apre = fast ? (unsigned short*)((char*)d_ws + offApre) : nullptr;

  enc_pack_w   <<<(N4H * KDIM) / 256, 256, 0, stream>>>(Wih, Whh, Wp);
  enc_bias     <<<4, 256, 0, stream>>>(bih, bhh, biasC);
  enc_attn_win <<<BB, 256, 0, stream>>>(x, Wattn, battn, outW, Apre);
  if (fast)
    enc_recurrent_async<<<BB / 16, 256, 0, stream>>>(Apre, Wp, biasC, outEnc);
  else
    enc_recurrent<<<BB / 16, 256, 0, stream>>>(outW, Wp, biasC, outEnc);
}
